// SMTA_50663434224271
// MI455X (gfx1250) — matrix operand order repaired and checked by reading, not compiled, not run on hardware
//
#include <hip/hip_runtime.h>

typedef __attribute__((ext_vector_type(16))) _Float16 v16h;
typedef __attribute__((ext_vector_type(8)))  float    v8f;

#define EPS_BN 1e-5f

struct P {
  const float *x;
  const float *n1g, *n1b, *n2g, *n2b;
  const float *p1w, *p1b;
  const float *c0w, *c0b;
  const float *cspw, *cspb;
  const float *c1w, *c1b, *c2w, *c2b;
  const float *fc1w, *f1g, *f1b, *fc2w;
  const float *cw, *cb;
  const float *p2w, *p2b;
  const float *mf1w, *mf1b, *mdww, *mdwb, *mf2w, *mf2b;
  const float *ls1, *ls2, *temp;
  const float *qkvw, *qkvb, *qdww, *qdwb;
  const float *pow_, *pob;
  const float *a1, *a2, *a3, *a4;
  const float *linw, *linb;
};

// Packed-weight LDS layout (floats)
enum {
  O_N1G = 0, O_N1B = 8, O_N2G = 16, O_N2B = 24,
  O_P1W = 32, O_P1B = 96,
  O_C0W = 104, O_C0B = 176,
  O_CSPW = 184, O_CSPB = 384,
  O_C1W = 392, O_C1B = 424, O_C2W = 428, O_C2B = 460,
  O_FC1W = 464, O_F1G = 720, O_F1B = 752,
  O_FC2W = 784,
  O_CW = 1040, O_CB = 1072,
  O_P2W = 1080, O_P2B = 1144,
  O_MF1W = 1152, O_MF1B = 1216,
  O_MDWW = 1224, O_MDWB = 1296,
  O_MF2W = 1304, O_MF2B = 1368,
  O_LS1 = 1376, O_LS2 = 1384, O_TEMP = 1392,
  O_QKVW = 1396, O_QKVB = 1588,
  O_QDWW = 1612, O_QDWB = 1828,
  O_POW = 1852, O_POB = 1916,
  O_A = 1924,
  O_LINB = 1928,
  W_TOTAL = 1992
};
// Workspace layout (float offsets): [0..95] stats, [128..2119] packed params,
// [2176..18559] lin_w as f16 (32768 halves), [18560..] x1 cache (ns*512 floats)
#define WS_PACK 128
#define WS_BH   2176
#define WS_X1   18560

__device__ __forceinline__ float gelu_f(float v) {
  return 0.5f * v * (1.0f + erff(v * 0.70710678118654752f));
}

// Branchless depthwise 3x3, pad 1: unconditional LDS read at wrapped (valid)
// address, zero-select on the value -> ds_load + v_cndmask + v_fmac, no branches.
__device__ __forceinline__ float dw3x3(const float* plane, const float* w, int i, int j) {
  float acc = 0.f;
#pragma unroll
  for (int di = 0; di < 3; ++di)
#pragma unroll
    for (int dj = 0; dj < 3; ++dj) {
      int ii = i + di - 1, jj = j + dj - 1;
      bool ok = ((unsigned)ii < 8u) && ((unsigned)jj < 8u);
      float v = plane[((ii & 7) << 3) | (jj & 7)];
      acc += w[di * 3 + dj] * (ok ? v : 0.f);
    }
  return acc;
}

// A fragment for V_WMMA_F32_16X16X32_F16 (contiguous k order: k = 16*khalf + s, matching B).
__device__ __forceinline__ v16h load_afrag(const float* Arow, int kb, int khalf) {
  v16h Af;
#pragma unroll
  for (int r = 0; r < 8; ++r) {
    int k0 = kb + 2 * r + 16 * khalf;
    Af[2 * r]     = (_Float16)Arow[k0];
    Af[2 * r + 1] = (_Float16)Arow[k0 + 1];
  }
  return Af;
}

// ---------------- Pack all small params into contiguous ws region ----------------
__global__ __launch_bounds__(256) void k_pack(P p, float* dst) {
  const float* srcs[42] = {
    p.n1g, p.n1b, p.n2g, p.n2b, p.p1w, p.p1b, p.c0w, p.c0b, p.cspw, p.cspb,
    p.c1w, p.c1b, p.c2w, p.c2b, p.fc1w, p.f1g, p.f1b, p.fc2w, p.cw, p.cb,
    p.p2w, p.p2b, p.mf1w, p.mf1b, p.mdww, p.mdwb, p.mf2w, p.mf2b, p.ls1, p.ls2,
    p.temp, p.qkvw, p.qkvb, p.qdww, p.qdwb, p.pow_, p.pob, p.a1, p.a2, p.a3,
    p.a4, p.linb };
  const int sz[42] = { 8, 8, 8, 8, 64, 8, 72, 8, 200, 8,
                       32, 4, 32, 4, 256, 32, 32, 256, 32, 8,
                       64, 8, 64, 8, 72, 8, 64, 8, 8, 8,
                       4, 192, 24, 216, 24, 64, 8, 1, 1, 1,
                       1, 64 };
  int off = 0;
  for (int g = 0; g < 42; ++g) {
    for (int t = threadIdx.x; t < sz[g]; t += blockDim.x) dst[off + t] = srcs[g][t];
    off += sz[g];
  }
}

// ---------------- Pre-convert lin_w to f16 (B operand) ----------------
__global__ __launch_bounds__(256) void k_cvt_b(const float* __restrict__ linw,
                                               _Float16* __restrict__ bh) {
  int idx = blockIdx.x * blockDim.x + threadIdx.x;
  if (idx < 64 * 512) bh[idx] = (_Float16)linw[idx];
}

// ---------------- Pass 1: batch stats of x (8 channels) ----------------
__global__ __launch_bounds__(256) void k_stats_x(const float* __restrict__ x,
                                                 float* ws, int ns) {
  __shared__ float acc[16];
  if (threadIdx.x < 16) acc[threadIdx.x] = 0.f;
  __syncthreads();
  float s[8] = {}, q[8] = {};
  const int total = ns * 64;
  for (int idx = blockIdx.x * blockDim.x + threadIdx.x; idx < total;
       idx += gridDim.x * blockDim.x) {
    const float* xp = x + (size_t)(idx >> 6) * 512 + (idx & 63);
#pragma unroll
    for (int c = 0; c < 8; ++c) { float v = xp[c * 64]; s[c] += v; q[c] += v * v; }
  }
#pragma unroll
  for (int c = 0; c < 8; ++c) {
    float sv = s[c], qv = q[c];
    for (int off = 16; off > 0; off >>= 1) {
      sv += __shfl_xor(sv, off);
      qv += __shfl_xor(qv, off);
    }
    if ((threadIdx.x & 31) == 0) { atomicAdd(&acc[c], sv); atomicAdd(&acc[8 + c], qv); }
  }
  __syncthreads();
  if (threadIdx.x < 16) atomicAdd(&ws[threadIdx.x], acc[threadIdx.x]);
}

// ---------------- Shared front: x -> h1 -> a -> t1a/t2a -> t1,t2, ch ----------------
// smlS: [0..7]=ch*64, [8..39]=z, [40..47]=|q|^2, [48..55]=|k|^2, [56..71]=q.k, [72..79]=logits
__device__ __forceinline__ void front(const float* xglob, const float* W,
                                      const float* ws, float nInv1,
                                      int s, int l, float* bufS, float* smlS,
                                      float* X, float* H1, float* A,
                                      float* T1A, float* T2A,
                                      float* T1, float* T2, float& AVG) {
  const int i = l >> 3, j = l & 7;
  for (int k = l; k < 96; k += 64) smlS[k] = 0.f;
  const float* xg = xglob + (size_t)s * 512 + l;
#pragma unroll
  for (int c = 0; c < 8; ++c) {
    float xv = xg[c * 64];
    X[c] = xv;
    float m = ws[c] * nInv1;
    float v = ws[8 + c] * nInv1 - m * m;
    H1[c] = (xv - m) * rsqrtf(v + EPS_BN) * W[O_N1G + c] + W[O_N1B + c];
  }
#pragma unroll
  for (int o = 0; o < 8; ++o) {
    float acc = W[O_P1B + o];
#pragma unroll
    for (int c = 0; c < 8; ++c) acc += W[O_P1W + o * 8 + c] * H1[c];
    A[o] = gelu_f(acc);
  }
#pragma unroll
  for (int c = 0; c < 8; ++c) bufS[c * 64 + l] = A[c];
  __syncthreads();
#pragma unroll
  for (int c = 0; c < 8; ++c)
    T1A[c] = W[O_C0B + c] + dw3x3(bufS + c * 64, W + O_C0W + c * 9, i, j);
  __syncthreads();
#pragma unroll
  for (int c = 0; c < 8; ++c) bufS[c * 64 + l] = T1A[c];
  __syncthreads();
#pragma unroll
  for (int c = 0; c < 8; ++c) {
    float acc = W[O_CSPB + c];
#pragma unroll
    for (int di = 0; di < 5; ++di)
#pragma unroll
      for (int dj = 0; dj < 5; ++dj) {
        int ii = i + 2 * di - 4, jj = j + 2 * dj - 4;
        bool ok = ((unsigned)ii < 8u) && ((unsigned)jj < 8u);
        float v = bufS[c * 64 + (((ii & 7) << 3) | (jj & 7))];
        acc += W[O_CSPW + c * 25 + di * 5 + dj] * (ok ? v : 0.f);
      }
    T2A[c] = acc;
  }
  __syncthreads();  // bufS reads complete
  float ssum = 0.f;
#pragma unroll
  for (int u = 0; u < 4; ++u) {
    float a1 = W[O_C1B + u], a2 = W[O_C2B + u];
#pragma unroll
    for (int c = 0; c < 8; ++c) {
      a1 += W[O_C1W + u * 8 + c] * T1A[c];
      a2 += W[O_C2W + u * 8 + c] * T2A[c];
    }
    T1[u] = a1; T2[u] = a2; ssum += a1 + a2;
  }
  AVG = ssum * 0.125f;
#pragma unroll
  for (int u = 0; u < 4; ++u) {
    atomicAdd(&smlS[u], T1[u]);
    atomicAdd(&smlS[4 + u], T2[u]);
  }
  __syncthreads();  // ch ready: smlS[c]/64
}

// ---------------- Shared mid: fc1bn -> fc2 softmax -> gate -> proj2 -> x1 ----------------
__device__ __forceinline__ void mid(const float* W, const float* ws, float nsInv,
                                    int l, float* smlS,
                                    const float* X, const float* H1, const float* A,
                                    const float* T1, const float* T2, float AVG,
                                    float* X1) {
  if (l < 32) {
    float pz = 0.f;
#pragma unroll
    for (int c = 0; c < 8; ++c) pz += W[O_FC1W + l * 8 + c] * (smlS[c] * 0.015625f);
    float m = ws[16 + l] * nsInv;
    float v = ws[48 + l] * nsInv - m * m;
    float z = (pz - m) * rsqrtf(v + EPS_BN) * W[O_F1G + l] + W[O_F1B + l];
    smlS[8 + l] = fmaxf(z, 0.f);
  }
  __syncthreads();
  // fc2 logits computed once per sample (8 threads), broadcast via LDS
  if (l < 8) {
    float lg = 0.f;
#pragma unroll
    for (int kk = 0; kk < 32; ++kk) lg += W[O_FC2W + l * 32 + kk] * smlS[8 + kk];
    smlS[72 + l] = lg;
  }
  __syncthreads();
  float a1u[4], a2u[4];
#pragma unroll
  for (int u = 0; u < 4; ++u) {
    float l0 = smlS[72 + u], l1 = smlS[72 + 4 + u];
    float mx = fmaxf(l0, l1);
    float e0 = expf(l0 - mx), e1 = expf(l1 - mx);
    float inv = 1.f / (e0 + e1);
    a1u[u] = e0 * inv; a2u[u] = e1 * inv;
  }
  float s4[4];
#pragma unroll
  for (int u = 0; u < 4; ++u) s4[u] = (T1[u] * a1u[u] + T2[u] * a2u[u]) * AVG;
  float AG[8];
#pragma unroll
  for (int c = 0; c < 8; ++c) {
    float g = W[O_CB + c];
#pragma unroll
    for (int u = 0; u < 4; ++u) g += W[O_CW + c * 4 + u] * s4[u];
    AG[c] = A[c] / (1.f + expf(-g));
  }
#pragma unroll
  for (int o = 0; o < 8; ++o) {
    float ao = W[O_P2B + o];
#pragma unroll
    for (int c = 0; c < 8; ++c) ao += W[O_P2W + o * 8 + c] * AG[c];
    ao += H1[o];
    X1[o] = X[o] + W[O_LS1 + o] * ao;
  }
}

#define LOAD_W()                                                        \
  __shared__ float W[W_TOTAL];                                          \
  for (int k = threadIdx.x; k < W_TOTAL; k += blockDim.x)               \
    W[k] = wsp[k];                                                      \
  __syncthreads();

// ---------------- Pass 2: accumulate pre_z batch stats (32 channels) ----------------
__global__ __launch_bounds__(1024) void k_pass2(const float* __restrict__ xglob,
                                                float* ws, int ns) {
  __shared__ float buf[16][512];
  __shared__ float sml[16][96];
  __shared__ float sacc[64];
  const float* wsp = ws + WS_PACK;
  const int tid = threadIdx.x;
  if (tid < 64) sacc[tid] = 0.f;
  LOAD_W();
  const int sl = tid >> 6, l = tid & 63;
  const int s = blockIdx.x * 16 + sl;
  const float nInv1 = 1.f / ((float)ns * 64.f);
  float X[8], H1[8], A[8], T1A[8], T2A[8], T1[4], T2[4], AVG;
  front(xglob, W, ws, nInv1, s, l, buf[sl], sml[sl], X, H1, A, T1A, T2A, T1, T2, AVG);
  if (l < 32) {
    float pz = 0.f;
#pragma unroll
    for (int c = 0; c < 8; ++c) pz += W[O_FC1W + l * 8 + c] * (sml[sl][c] * 0.015625f);
    atomicAdd(&sacc[l], pz);
    atomicAdd(&sacc[32 + l], pz * pz);
  }
  __syncthreads();
  if (tid < 32) atomicAdd(&ws[16 + tid], sacc[tid]);
  else if (tid < 64) atomicAdd(&ws[48 + (tid - 32)], sacc[tid]);
}

// ---------------- Pass 3: x1 batch stats; optionally cache x1 to ws ----------------
template <bool STORE>
__global__ __launch_bounds__(1024) void k_pass3(const float* __restrict__ xglob,
                                                float* ws, float* __restrict__ x1c,
                                                int ns) {
  __shared__ float buf[16][512];
  __shared__ float sml[16][96];
  __shared__ float sacc[16];
  const float* wsp = ws + WS_PACK;
  const int tid = threadIdx.x;
  if (tid < 16) sacc[tid] = 0.f;
  LOAD_W();
  const int sl = tid >> 6, l = tid & 63;
  const int s = blockIdx.x * 16 + sl;
  const float nInv1 = 1.f / ((float)ns * 64.f);
  const float nsInv = 1.f / (float)ns;
  float X[8], H1[8], A[8], T1A[8], T2A[8], T1[4], T2[4], AVG, X1[8];
  front(xglob, W, ws, nInv1, s, l, buf[sl], sml[sl], X, H1, A, T1A, T2A, T1, T2, AVG);
  mid(W, ws, nsInv, l, sml[sl], X, H1, A, T1, T2, AVG, X1);
  if constexpr (STORE) {
    float* cg = x1c + (size_t)s * 512 + l;
#pragma unroll
    for (int c = 0; c < 8; ++c) cg[c * 64] = X1[c];
  }
#pragma unroll
  for (int c = 0; c < 8; ++c) {
    float sv = X1[c], qv = sv * sv;
    for (int off = 16; off > 0; off >>= 1) {
      sv += __shfl_xor(sv, off);
      qv += __shfl_xor(qv, off);
    }
    if ((tid & 31) == 0) { atomicAdd(&sacc[c], sv); atomicAdd(&sacc[8 + c], qv); }
  }
  __syncthreads();
  if (tid < 16) atomicAdd(&ws[80 + tid], sacc[tid]);
}

// ---------------- Pass 4: (cached) forward tail + WMMA GEMM with lin_w ----------------
template <bool CACHED>
__global__ __launch_bounds__(1024) void k_pass4(const float* __restrict__ xglob,
                                                const float* __restrict__ x1c,
                                                const float* __restrict__ linw,
                                                const _Float16* __restrict__ bh,
                                                float* ws, float* out, int ns) {
  __shared__ float buf[16][512];   // stage planes, then 16x512 A-tile for WMMA
  __shared__ float sml[16][96];
  const float* wsp = ws + WS_PACK;
  const int tid = threadIdx.x;
  LOAD_W();
  const int sl = tid >> 6, l = tid & 63;
  const int i = l >> 3, j = l & 7;
  const int s = blockIdx.x * 16 + sl;
  float* bufS = buf[sl];
  float* smlS = sml[sl];
  const float nInv1 = 1.f / ((float)ns * 64.f);
  const float nsInv = 1.f / (float)ns;

  // Warm L2/WGP$ with the lin_w rows this wave's GEMM will consume.
  if (tid < 128) {
    const int lane = tid & 31;
    const size_t row = (size_t)((tid >> 5) * 16 + (lane & 15)) * 512 + 16 * (lane >> 4);
    if (bh) {
      __builtin_prefetch(bh + row, 0, 3);
      __builtin_prefetch(bh + row + 256, 0, 3);
    } else {
#pragma unroll
      for (int pf = 0; pf < 8; ++pf) __builtin_prefetch(linw + row + pf * 64, 0, 3);
    }
  }

  float X[8], X1[8];
  if constexpr (CACHED) {
    for (int k2 = l; k2 < 96; k2 += 64) smlS[k2] = 0.f;
    const float* xg = xglob + (size_t)s * 512 + l;
    const float* cg = x1c + (size_t)s * 512 + l;
#pragma unroll
    for (int c = 0; c < 8; ++c) { X[c] = xg[c * 64]; X1[c] = cg[c * 64]; }
  } else {
    float H1[8], A[8], T1A[8], T2A[8], T1[4], T2[4], AVG;
    front(xglob, W, ws, nInv1, s, l, bufS, smlS, X, H1, A, T1A, T2A, T1, T2, AVG);
    mid(W, ws, nsInv, l, smlS, X, H1, A, T1, T2, AVG, X1);
  }

  // ---- BN2 + MLP ----
  float H2[8], MM[8];
#pragma unroll
  for (int c = 0; c < 8; ++c) {
    float m = ws[80 + c] * nInv1;
    float v = ws[88 + c] * nInv1 - m * m;
    H2[c] = (X1[c] - m) * rsqrtf(v + EPS_BN) * W[O_N2G + c] + W[O_N2B + c];
  }
#pragma unroll
  for (int c = 0; c < 8; ++c) {
    float acc = W[O_MF1B + c];
#pragma unroll
    for (int d = 0; d < 8; ++d) acc += W[O_MF1W + c * 8 + d] * H2[d];
    MM[c] = acc;
  }
  __syncthreads();  // (CACHED path: ensure W/sml init visible before staging)
#pragma unroll
  for (int c = 0; c < 8; ++c) bufS[c * 64 + l] = MM[c];
  __syncthreads();
  float MD[8];
#pragma unroll
  for (int c = 0; c < 8; ++c)
    MD[c] = gelu_f(W[O_MDWB + c] + dw3x3(bufS + c * 64, W + O_MDWW + c * 9, i, j));
  __syncthreads();
  float OS1[8], Y[8];
#pragma unroll
  for (int c = 0; c < 8; ++c) {
    float acc = W[O_MF2B + c];
#pragma unroll
    for (int d = 0; d < 8; ++d) acc += W[O_MF2W + c * 8 + d] * MD[d];
    OS1[c] = X1[c] + W[O_LS2 + c] * acc;
    Y[c] = OS1[c] + X[c];
  }

  // ---- qkv = dw3x3(conv1x1(y)), 24 channels in 3 rounds of 8 ----
  float QK[24], QD[24];
#pragma unroll
  for (int o = 0; o < 24; ++o) {
    float acc = W[O_QKVB + o];
#pragma unroll
    for (int c = 0; c < 8; ++c) acc += W[O_QKVW + o * 8 + c] * Y[c];
    QK[o] = acc;
  }
  for (int g = 0; g < 3; ++g) {
#pragma unroll
    for (int c = 0; c < 8; ++c) bufS[c * 64 + l] = QK[g * 8 + c];
    __syncthreads();
#pragma unroll
    for (int c = 0; c < 8; ++c) {
      int ch = g * 8 + c;
      QD[ch] = W[O_QDWB + ch] + dw3x3(bufS + c * 64, W + O_QDWW + ch * 9, i, j);
    }
    __syncthreads();
  }

  // ---- l2norm + q.k dots (smlS[40..71] pre-zeroed) ----
#pragma unroll
  for (int hc = 0; hc < 8; ++hc) {
    atomicAdd(&smlS[40 + hc], QD[hc] * QD[hc]);          // |q|^2
    atomicAdd(&smlS[48 + hc], QD[8 + hc] * QD[8 + hc]);  // |k|^2
  }
#pragma unroll
  for (int h = 0; h < 4; ++h)
#pragma unroll
    for (int c2 = 0; c2 < 2; ++c2)
#pragma unroll
      for (int d = 0; d < 2; ++d)
        atomicAdd(&smlS[56 + h * 4 + c2 * 2 + d], QD[h * 2 + c2] * QD[8 + h * 2 + d]);
  __syncthreads();

  // ---- top-1 "softmax" attention: out = (a1+a2+a3+a4) * v[argmax] ----
  const float Asc = W[O_A] + W[O_A + 1] + W[O_A + 2] + W[O_A + 3];
  float OC[8];
#pragma unroll
  for (int h = 0; h < 4; ++h) {
    float nk0 = fmaxf(sqrtf(smlS[48 + h * 2 + 0]), 1e-12f);
    float nk1 = fmaxf(sqrtf(smlS[48 + h * 2 + 1]), 1e-12f);
#pragma unroll
    for (int c2 = 0; c2 < 2; ++c2) {
      float nq = fmaxf(sqrtf(smlS[40 + h * 2 + c2]), 1e-12f);
      float att0 = W[O_TEMP + h] * smlS[56 + h * 4 + c2 * 2 + 0] / (nq * nk0);
      float att1 = W[O_TEMP + h] * smlS[56 + h * 4 + c2 * 2 + 1] / (nq * nk1);
      int dsel = (att1 > att0) ? 1 : 0;
      OC[h * 2 + c2] = Asc * QD[16 + h * 2 + dsel];
    }
  }
  float FIN[8];
#pragma unroll
  for (int o = 0; o < 8; ++o) {
    float acc = W[O_POB + o];
#pragma unroll
    for (int c = 0; c < 8; ++c) acc += W[O_POW + o * 8 + c] * OC[c];
    FIN[o] = acc + OS1[o];
  }
  __syncthreads();
#pragma unroll
  for (int c = 0; c < 8; ++c) bufS[c * 64 + l] = FIN[c];  // A-tile row sl, col c*64+l
  __syncthreads();

  // ---- final linear via WMMA: D(16x64) = A(16x512) x lin_w^T(512x64) + lin_b ----
  if (tid < 128) {
    const int lane = tid & 31;
    const int w = tid >> 5;       // N-tile 0..3
    const int khalf = lane >> 4;  // lane half -> K sub-block / D row half
    const int lm = lane & 15;     // A: row M; B,D: col N within tile
    const int nb = w * 16;
    v8f Cacc = {};
    const float* Arow = buf[lm];
    if (bh) {
      const _Float16* Bb = bh + (size_t)(nb + lm) * 512 + 16 * khalf;
#pragma unroll
      for (int kb = 0; kb < 512; kb += 32) {
        v16h Af = load_afrag(Arow, kb, khalf), Bf;
#pragma unroll
        for (int e = 0; e < 16; ++e) Bf[e] = Bb[kb + e];
        Cacc = __builtin_amdgcn_wmma_f32_16x16x32_f16(
            false, Af, false, Bf, (short)0, Cacc, false, false);
      }
    } else {
      const float* Bb = linw + (size_t)(nb + lm) * 512 + 16 * khalf;
#pragma unroll
      for (int kb = 0; kb < 512; kb += 32) {
        v16h Af = load_afrag(Arow, kb, khalf), Bf;
#pragma unroll
        for (int e = 0; e < 16; ++e) Bf[e] = (_Float16)Bb[kb + e];
        Cacc = __builtin_amdgcn_wmma_f32_16x16x32_f16(
            false, Af, false, Bf, (short)0, Cacc, false, false);
      }
    }
    const int sBase = blockIdx.x * 16;
    const float bias = W[O_LINB + nb + lm];
#pragma unroll
    for (int iv = 0; iv < 8; ++iv) {
      int Mrow = iv + 8 * khalf;  // D: VGPR iv -> sample row iv (+8 for upper lanes)
      out[(size_t)(sBase + Mrow) * 64 + nb + lm] = Cacc[iv] + bias;
    }
  }
}

extern "C" void kernel_launch(void* const* d_in, const int* in_sizes, int n_in,
                              void* d_out, int out_size, void* d_ws, size_t ws_size,
                              hipStream_t stream) {
  P p;
  int k = 0;
  p.x    = (const float*)d_in[k++];
  p.n1g  = (const float*)d_in[k++];  p.n1b  = (const float*)d_in[k++];
  p.n2g  = (const float*)d_in[k++];  p.n2b  = (const float*)d_in[k++];
  p.p1w  = (const float*)d_in[k++];  p.p1b  = (const float*)d_in[k++];
  p.c0w  = (const float*)d_in[k++];  p.c0b  = (const float*)d_in[k++];
  p.cspw = (const float*)d_in[k++];  p.cspb = (const float*)d_in[k++];
  p.c1w  = (const float*)d_in[k++];  p.c1b  = (const float*)d_in[k++];
  p.c2w  = (const float*)d_in[k++];  p.c2b  = (const float*)d_in[k++];
  p.fc1w = (const float*)d_in[k++];  p.f1g  = (const float*)d_in[k++];
  p.f1b  = (const float*)d_in[k++];  p.fc2w = (const float*)d_in[k++];
  p.cw   = (const float*)d_in[k++];  p.cb   = (const float*)d_in[k++];
  p.p2w  = (const float*)d_in[k++];  p.p2b  = (const float*)d_in[k++];
  p.mf1w = (const float*)d_in[k++];  p.mf1b = (const float*)d_in[k++];
  p.mdww = (const float*)d_in[k++];  p.mdwb = (const float*)d_in[k++];
  p.mf2w = (const float*)d_in[k++];  p.mf2b = (const float*)d_in[k++];
  p.ls1  = (const float*)d_in[k++];  p.ls2  = (const float*)d_in[k++];
  p.temp = (const float*)d_in[k++];
  p.qkvw = (const float*)d_in[k++];  p.qkvb = (const float*)d_in[k++];
  p.qdww = (const float*)d_in[k++];  p.qdwb = (const float*)d_in[k++];
  p.pow_ = (const float*)d_in[k++];  p.pob  = (const float*)d_in[k++];
  p.a1   = (const float*)d_in[k++];  p.a2   = (const float*)d_in[k++];
  p.a3   = (const float*)d_in[k++];  p.a4   = (const float*)d_in[k++];
  p.linw = (const float*)d_in[k++];  p.linb = (const float*)d_in[k++];

  const int ns = in_sizes[0] / 512;  // 32768 samples
  float* ws = (float*)d_ws;

  // Workspace-gated fast paths (deterministic: ws_size is fixed per run)
  const size_t need_med = (size_t)WS_X1 * sizeof(float);
  const size_t need_big = ((size_t)WS_X1 + (size_t)ns * 512) * sizeof(float);
  _Float16* bh = (ws_size >= need_med) ? (_Float16*)(ws + WS_BH) : nullptr;
  float* x1c   = (ws_size >= need_big) ? (ws + WS_X1) : nullptr;

  hipMemsetAsync(d_ws, 0, 96 * sizeof(float), stream);
  k_pack<<<1, 256, 0, stream>>>(p, ws + WS_PACK);
  if (bh) k_cvt_b<<<128, 256, 0, stream>>>(p.linw, bh);
  k_stats_x<<<256, 256, 0, stream>>>(p.x, ws, ns);
  k_pass2<<<ns / 16, 1024, 0, stream>>>(p.x, ws, ns);
  if (x1c) {
    k_pass3<true><<<ns / 16, 1024, 0, stream>>>(p.x, ws, x1c, ns);
    k_pass4<true><<<ns / 16, 1024, 0, stream>>>(p.x, x1c, p.linw, bh, ws,
                                                (float*)d_out, ns);
  } else {
    k_pass3<false><<<ns / 16, 1024, 0, stream>>>(p.x, ws, nullptr, ns);
    k_pass4<false><<<ns / 16, 1024, 0, stream>>>(p.x, nullptr, p.linw, bh, ws,
                                                 (float*)d_out, ns);
  }
}
